// AttentionMechanism_58574763983667
// MI455X (gfx1250) — compile-verified, matrix operand order checked by reading, not run on hardware
//
#include <hip/hip_runtime.h>

#define Bsz 256
#define Ssz 1024
#define Esz 512
#define Dsz 512
#define Hsz 512

typedef __attribute__((ext_vector_type(16))) __bf16          v16bf;
typedef __attribute__((ext_vector_type(8)))  float           v8f;
typedef __attribute__((ext_vector_type(8)))  int             v8i;

__device__ __forceinline__ unsigned short f2bf(float f) {
    unsigned int u = __float_as_uint(f);
    u += 0x7fffu + ((u >> 16) & 1u);      // round-to-nearest-even
    return (unsigned short)(u >> 16);
}

// Async DMA of 64 bytes (4 x b128) global -> LDS, tracked by ASYNCcnt.
// IOFFSET is applied to BOTH the LDS and global addresses (ISA 10.7).
__device__ __forceinline__ void async_cp64(unsigned lds_off,
                                           const unsigned short* src) {
    asm volatile(
        "global_load_async_to_lds_b128 %0, %1, off\n\t"
        "global_load_async_to_lds_b128 %0, %1, off offset:16\n\t"
        "global_load_async_to_lds_b128 %0, %1, off offset:32\n\t"
        "global_load_async_to_lds_b128 %0, %1, off offset:48"
        :: "v"(lds_off), "v"(src) : "memory");
}

__device__ __forceinline__ void wait_async0() {
    asm volatile("s_wait_asynccnt 0" ::: "memory");
}

// ---------------------------------------------------------------------------
// Kernel 1: repack W_enc [E,H] fp32 -> bf16 fragments in WMMA B layout.
// Fragment index = (((hc*16 + kc)*32) + lane)*16 + i
//   n = hc*16 + (lane&15)
//   kbase = kc*32 + (lane>=16 ? 8 : 0)
//   k = kbase + (i<8 ? i : i+8)        (elements 8..15 hold K+16..K+23)
// ---------------------------------------------------------------------------
__global__ __launch_bounds__(256) void pack_wenc(
    const float* __restrict__ wenc, unsigned short* __restrict__ dst)
{
    int idx  = blockIdx.x * 256 + threadIdx.x;     // 0 .. 512*512-1
    int i    = idx & 15;
    int lane = (idx >> 4) & 31;
    int kc   = (idx >> 9) & 15;
    int hc   = idx >> 13;
    int n    = hc * 16 + (lane & 15);
    int kb   = kc * 32 + ((lane >> 4) ? 8 : 0);
    int k    = kb + (i < 8 ? i : i + 8);
    dst[idx] = f2bf(wenc[k * Hsz + n]);
}

// ---------------------------------------------------------------------------
// Kernel 2: addv[b,h] = decoder_state[b] @ W_dec + b_dec + b_enc
// ---------------------------------------------------------------------------
__global__ __launch_bounds__(256) void addvec(
    const float* __restrict__ dec, const float* __restrict__ wdec,
    const float* __restrict__ bdec, const float* __restrict__ benc,
    float* __restrict__ out)
{
    __shared__ float sdec[Dsz];
    int b = blockIdx.x, t = threadIdx.x;
    sdec[t]       = dec[b * Dsz + t];
    sdec[t + 256] = dec[b * Dsz + t + 256];
    __syncthreads();
    float a0 = 0.f, a1 = 0.f;
    for (int d = 0; d < Dsz; ++d) {
        float dv = sdec[d];
        const float* wr = wdec + d * Hsz;
        a0 += dv * wr[t];
        a1 += dv * wr[t + 256];
    }
    out[b * Hsz + t]       = a0 + bdec[t]       + benc[t];
    out[b * Hsz + t + 256] = a1 + bdec[t + 256] + benc[t + 256];
}

// ---------------------------------------------------------------------------
// Kernel 3: fused attention. One workgroup (8 waves) per batch.
// A tile (16 rows x K=512, bf16) register-resident per wave;
// B tiles (16 KB per h-chunk) DMA'd into LDS asynchronously, double-buffered.
// ---------------------------------------------------------------------------
__global__ __launch_bounds__(256) void attn_main(
    const float* __restrict__ enc,            // [B,S,E]
    const unsigned char* __restrict__ mask,   // [B,S] bool
    const unsigned short* __restrict__ wencP, // packed bf16 B fragments
    const float* __restrict__ addv,           // [B,H]
    const float* __restrict__ watt,           // [H]
    const float* __restrict__ batt,           // [1]
    const float* __restrict__ wctx,           // [E,D]
    const float* __restrict__ bctx,           // [D]
    float* __restrict__ out)                  // [B*D] ctx ++ [B*S] weights
{
    __shared__ unsigned short s_b[2][16 * 512]; // 2 x 16KB B tiles
    __shared__ float s_scores[Ssz];
    __shared__ float s_addv[Hsz];
    __shared__ float s_watt[Hsz];
    __shared__ float s_red[256];
    __shared__ float s_ctx[Esz];

    const int b = blockIdx.x, t = threadIdx.x;
    const int wave = t >> 5, lane = t & 31;
    const int laneHi = lane >> 4, ln = lane & 15;

    // LDS byte offsets of the two staging buffers for this thread's 64B slice
    // (generic LDS pointers carry the LDS offset in their low 32 bits).
    const unsigned lds0 =
        (unsigned)(uintptr_t)(&s_b[0][0]) + (unsigned)t * 64u;
    const unsigned lds1 = lds0 + 16u * 512u * 2u;

    s_addv[t]       = addv[b * Hsz + t];
    s_addv[t + 256] = addv[b * Hsz + t + 256];
    s_watt[t]       = watt[t];
    s_watt[t + 256] = watt[t + 256];
    const float battv = batt[0];

    // ---------------- Phase A: scores via bf16 WMMA -----------------------
    for (int sIter = 0; sIter < 8; ++sIter) {
        const int s0  = sIter * 128 + wave * 16;   // tile base row in batch
        const int row = s0 + ln;                   // this lane's A row
        const float* arow = enc + ((size_t)b * Ssz + row) * Esz;

        // kick off async DMA of the hc=0 B tile (overlaps the A convert)
        async_cp64(lds0, wencP + (size_t)t * 32);

        // load + convert the full A tile once: 16 fragments = 128 VGPRs
        v16bf afrag[16];
#pragma unroll
        for (int kc = 0; kc < 16; ++kc) {
            const int kb = kc * 32 + laneHi * 8;
            float4 a0 = *(const float4*)(arow + kb);
            float4 a1 = *(const float4*)(arow + kb + 4);
            float4 a2 = *(const float4*)(arow + kb + 16);
            float4 a3 = *(const float4*)(arow + kb + 20);
            v16bf af;
            af[0]  = (__bf16)a0.x; af[1]  = (__bf16)a0.y;
            af[2]  = (__bf16)a0.z; af[3]  = (__bf16)a0.w;
            af[4]  = (__bf16)a1.x; af[5]  = (__bf16)a1.y;
            af[6]  = (__bf16)a1.z; af[7]  = (__bf16)a1.w;
            af[8]  = (__bf16)a2.x; af[9]  = (__bf16)a2.y;
            af[10] = (__bf16)a2.z; af[11] = (__bf16)a2.w;
            af[12] = (__bf16)a3.x; af[13] = (__bf16)a3.y;
            af[14] = (__bf16)a3.z; af[15] = (__bf16)a3.w;
            afrag[kc] = af;
        }

        wait_async0();
        __syncthreads();

        float sacc[8];
#pragma unroll
        for (int r = 0; r < 8; ++r) sacc[r] = 0.f;

        for (int hc = 0; hc < 32; ++hc) {
            const int cur = hc & 1;
            // async-prefetch next B tile into the other buffer
            if (hc < 31) {
                async_cp64(cur ? lds0 : lds1,
                           wencP + (size_t)(hc + 1) * (16 * 512) +
                                   (size_t)t * 32);
            }

            v8f acc = {};
#pragma unroll
            for (int kc = 0; kc < 16; ++kc) {
                const v8i braw =
                    *(const v8i*)(&s_b[cur][(kc * 32 + lane) * 16]);
                v16bf bvf = __builtin_bit_cast(v16bf, braw);
                acc = __builtin_amdgcn_wmma_f32_16x16x32_bf16(
                    false, afrag[kc], false, bvf, (short)0, acc, false, false);
            }

            // acc[r] = enc_p[m, n], m = r + 8*laneHi, n = hc*16 + ln
            const int h = hc * 16 + ln;
            const float av = s_addv[h], ww = s_watt[h];
#pragma unroll
            for (int r = 0; r < 8; ++r) {
                float tv = tanhf(acc[r] + av);
                sacc[r] += tv * ww;
            }

            wait_async0();
            __syncthreads();
        }

        // reduce across the 16 lanes of each half-wave (n dimension)
#pragma unroll
        for (int r = 0; r < 8; ++r) {
            for (int off = 1; off < 16; off <<= 1)
                sacc[r] += __shfl_xor(sacc[r], off, 32);
        }
        if (ln == 0) {
#pragma unroll
            for (int r = 0; r < 8; ++r) {
                int srow = s0 + r + 8 * laneHi;
                float v  = sacc[r] + battv;
                if (!mask[(size_t)b * Ssz + srow]) v = -__builtin_inff();
                s_scores[srow] = v;
            }
        }
    }
    __syncthreads();

    // ---------------- Phase B: softmax over S -----------------------------
    float m0 = -__builtin_inff();
#pragma unroll
    for (int i = 0; i < 4; ++i) m0 = fmaxf(m0, s_scores[t + i * 256]);
    s_red[t] = m0;
    __syncthreads();
    for (int stp = 128; stp > 0; stp >>= 1) {
        if (t < stp) s_red[t] = fmaxf(s_red[t], s_red[t + stp]);
        __syncthreads();
    }
    const float gmax = s_red[0];
    __syncthreads();
    float lsum = 0.f;
#pragma unroll
    for (int i = 0; i < 4; ++i) {
        float e = __expf(s_scores[t + i * 256] - gmax);
        s_scores[t + i * 256] = e;
        lsum += e;
    }
    s_red[t] = lsum;
    __syncthreads();
    for (int stp = 128; stp > 0; stp >>= 1) {
        if (t < stp) s_red[t] += s_red[t + stp];
        __syncthreads();
    }
    const float inv = 1.0f / s_red[0];
    __syncthreads();
#pragma unroll
    for (int i = 0; i < 4; ++i) {
        float w = s_scores[t + i * 256] * inv;
        s_scores[t + i * 256] = w;
        out[Bsz * Dsz + (size_t)b * Ssz + t + i * 256] = w;
    }
    __syncthreads();

    // ---------------- Phase C: context = sum_s w[s] * enc[b,s,:] ----------
    float cx = 0.f, cy = 0.f;
    const float* ebase = enc + (size_t)b * Ssz * Esz;
#pragma unroll 4
    for (int s = 0; s < Ssz; ++s) {
        float w  = s_scores[s];
        float2 v = *(const float2*)(ebase + (size_t)s * Esz + 2 * t);
        cx += w * v.x;
        cy += w * v.y;
    }
    s_ctx[2 * t]     = cx;
    s_ctx[2 * t + 1] = cy;
    __syncthreads();

    // ---------------- Phase D: ctx @ W_ctx + b_ctx ------------------------
    float o0 = 0.f, o1 = 0.f;
    for (int e = 0; e < Esz; ++e) {
        float c = s_ctx[e];
        const float* wr = wctx + e * Dsz;
        o0 += c * wr[t];
        o1 += c * wr[t + 256];
    }
    out[(size_t)b * Dsz + t]       = o0 + bctx[t];
    out[(size_t)b * Dsz + t + 256] = o1 + bctx[t + 256];
}

// ---------------------------------------------------------------------------
extern "C" void kernel_launch(void* const* d_in, const int* in_sizes, int n_in,
                              void* d_out, int out_size, void* d_ws, size_t ws_size,
                              hipStream_t stream)
{
    const float*         enc   = (const float*)d_in[0];
    const float*         dec   = (const float*)d_in[1];
    const unsigned char* msk   = (const unsigned char*)d_in[2];
    const float*         wenc  = (const float*)d_in[3];
    const float*         benc  = (const float*)d_in[4];
    const float*         wdec  = (const float*)d_in[5];
    const float*         bdec  = (const float*)d_in[6];
    const float*         watt  = (const float*)d_in[7];
    const float*         batt  = (const float*)d_in[8];
    const float*         wctx  = (const float*)d_in[9];
    const float*         bctx  = (const float*)d_in[10];
    float*               out   = (float*)d_out;

    unsigned short* ws_wencP = (unsigned short*)d_ws;                         // 512 KB
    float*          ws_addv  = (float*)((char*)d_ws + (size_t)Esz * Hsz * 2); // 512 KB

    pack_wenc<<<(Esz * Hsz) / 256, 256, 0, stream>>>(wenc, ws_wencP);
    addvec<<<Bsz, 256, 0, stream>>>(dec, wdec, bdec, benc, ws_addv);
    attn_main<<<Bsz, 256, 0, stream>>>(enc, msk, ws_wencP, ws_addv,
                                       watt, batt, wctx, bctx, out);
}
